// SGCNet_23828478558588
// MI455X (gfx1250) — matrix operand order repaired and checked by reading, not compiled, not run on hardware
//
#include <hip/hip_runtime.h>
#include <hip/hip_fp16.h>
#include <math.h>

typedef __attribute__((ext_vector_type(16))) _Float16 v16h;
typedef __attribute__((ext_vector_type(8)))  float    v8f;

#define N_NODES   100000
#define N_EDGES   1600000
#define N_FEAT    50
#define N_CLASSES 40
#define KPAD      64          // padded K for WMMA (2 k-steps of 32)
#define HSTRIDE   64          // padded row stride of h1

// ---------------- degree / normalization ----------------

__global__ void k_deg_init(float* __restrict__ deg) {
    int i = blockIdx.x * blockDim.x + threadIdx.x;
    if (i < N_NODES) deg[i] = 1.0f;   // self-loop contributes 1
}

__global__ void k_deg_accum(const int* __restrict__ col, float* __restrict__ deg) {
    int i = blockIdx.x * blockDim.x + threadIdx.x;
    int stride = gridDim.x * blockDim.x;
    for (; i < N_EDGES; i += stride) {
        atomicAdd(&deg[col[i]], 1.0f);
    }
}

__global__ void k_rsqrt_inplace(float* __restrict__ deg) {
    int i = blockIdx.x * blockDim.x + threadIdx.x;
    if (i < N_NODES) deg[i] = rsqrtf(deg[i]);   // deg >= 1 always (self-loop)
}

// ---------------- propagation hops ----------------

// hop-1 init: h0[n][f] = dinv[n]^2 * x[n][f]   (stride 50 -> stride 50)
__global__ void k_selfloop50(const float* __restrict__ hin,
                             const float* __restrict__ dinv,
                             float* __restrict__ hout) {
    int i = blockIdx.x * blockDim.x + threadIdx.x;
    if (i < N_NODES * N_FEAT) {
        int n = i / N_FEAT;
        float d = dinv[n];
        hout[i] = d * d * hin[i];
    }
}

// hop-2 init: h1[n][f] = dinv[n]^2 * h0[n][f] for f<50, 0 for 50..63 (stride 64 out)
__global__ void k_selfloop_pad(const float* __restrict__ hin,
                               const float* __restrict__ dinv,
                               float* __restrict__ hout) {
    int i = blockIdx.x * blockDim.x + threadIdx.x;
    if (i < N_NODES * HSTRIDE) {
        int n = i >> 6;           // / 64
        int f = i & 63;
        float d = dinv[n];
        hout[i] = (f < N_FEAT) ? d * d * hin[n * N_FEAT + f] : 0.0f;
    }
}

// one wave32 per edge: hout[col] += dinv[row]*dinv[col] * hin[row]
// (templated on output row stride: 50 for hop 1, 64 for hop 2)
template <int OSTRIDE>
__global__ void k_edge_hop(const int* __restrict__ row,
                           const int* __restrict__ col,
                           const float* __restrict__ dinv,
                           const float* __restrict__ hin,
                           float* __restrict__ hout) {
    int lane   = threadIdx.x & 31;
    int wave   = (blockIdx.x * blockDim.x + threadIdx.x) >> 5;
    int nwaves = (gridDim.x * blockDim.x) >> 5;
    for (int e = wave; e < N_EDGES; e += nwaves) {
        int r = row[e];
        int c = col[e];
        float norm = dinv[r] * dinv[c];
        const float* src = hin  + r * N_FEAT;
        float*       dst = hout + c * OSTRIDE;
        atomicAdd(&dst[lane], norm * src[lane]);           // feats 0..31
        int f2 = lane + 32;
        if (f2 < N_FEAT) {                                  // feats 32..49
            atomicAdd(&dst[f2], norm * src[f2]);
        }
    }
}

// ---------------- B-fragment pre-pack ----------------
// Pack W[50,40] (f32) into per-lane WMMA B fragments, zero-padded to K=64, N=48.
// Layout: bpack[((ks*3 + nt)*32 + lane)*16 + idx], idx'th f16 of the v16h fragment.
// k order used here (contiguous, same as the A loads): lane group g = lane>>4 covers K = ks*32 + g*16 + idx.
__global__ void k_pack_B(const float* __restrict__ W, _Float16* __restrict__ bpack) {
    int t = blockIdx.x * blockDim.x + threadIdx.x;   // 0 .. 6*32-1
    if (t >= 6 * 32) return;
    int lane = t & 31;
    int frag = t >> 5;          // ks*3 + nt
    int ks   = frag / 3;
    int nt   = frag % 3;
    int g    = lane >> 4;
    int colN = nt * 16 + (lane & 15);
    #pragma unroll
    for (int idx = 0; idx < 16; ++idx) {
        int k = ks * 32 + g * 16 + idx;
        float v = (k < N_FEAT && colN < N_CLASSES) ? W[k * N_CLASSES + colN] : 0.0f;
        bpack[t * 16 + idx] = (_Float16)v;
    }
}

// ---------------- dense linear via WMMA ----------------
// logits = h @ W + b ; h:[N,64] padded f32.  One wave per 16-row tile.
// All fragment loads are unconditional (padding handled upstream).

__global__ void k_linear_wmma(const float* __restrict__ h,
                              const v16h* __restrict__ bpack,
                              const float* __restrict__ bias,
                              float* __restrict__ out) {
    int lane = threadIdx.x & 31;
    int tile = blockIdx.x * (blockDim.x >> 5) + (threadIdx.x >> 5);
    if (tile >= N_NODES / 16) return;          // wave-uniform: EXEC all-1 at WMMA

    int l15 = lane & 15;
    int g   = lane >> 4;                       // lane half-group
    const float* hr = h + (tile * 16 + l15) * HSTRIDE;   // A row for this lane

    v8f c0 = {}, c1 = {}, c2 = {};

    #pragma unroll
    for (int ks = 0; ks < 2; ++ks) {
        // A fragment: 16-bit A 16x32 layout; unconditional float2 loads + pk cvt.
        v16h a;
        #pragma unroll
        for (int j = 0; j < 4; ++j) {
            float2 p0 = *(const float2*)(hr + ks * 32 + g * 16 + 2 * j);
            a[2 * j]     = (_Float16)p0.x;
            a[2 * j + 1] = (_Float16)p0.y;
            float2 p1 = *(const float2*)(hr + ks * 32 + g * 16 + 8 + 2 * j);
            a[8 + 2 * j]     = (_Float16)p1.x;
            a[8 + 2 * j + 1] = (_Float16)p1.y;
        }
        #pragma unroll
        for (int nt = 0; nt < 3; ++nt) {
            v16h bf = bpack[(ks * 3 + nt) * 32 + lane];   // 2x global_load_b128
            v8f& acc = (nt == 0) ? c0 : ((nt == 1) ? c1 : c2);
            acc = __builtin_amdgcn_wmma_f32_16x16x32_f16(
                      false, a, false, bf, (short)0, acc, false, false);
        }
    }

    // C/D layout: lanes 0-15 -> M=r (VGPR r), lanes 16-31 -> M=8+r ; N = lane&15
    #pragma unroll
    for (int nt = 0; nt < 3; ++nt) {
        int colN = nt * 16 + l15;
        if (colN >= N_CLASSES) continue;       // only nt==2, lanes l15>=8 skip
        v8f acc = (nt == 0) ? c0 : ((nt == 1) ? c1 : c2);
        float bv = bias[colN];
        #pragma unroll
        for (int r = 0; r < 8; ++r) {
            int m = g * 8 + r;
            out[(tile * 16 + m) * N_CLASSES + colN] = acc[r] + bv;
        }
    }
}

// ---------------- row-wise log-softmax (in-place, one wave per row) ----------------

__global__ void k_logsoftmax(float* __restrict__ out) {
    int lane = threadIdx.x & 31;
    int wave = (blockIdx.x * blockDim.x + threadIdx.x) >> 5;
    if (wave >= N_NODES) return;
    float* r = out + wave * N_CLASSES;

    float v0 = r[lane];                                         // cols 0..31
    bool  has2 = (lane < (N_CLASSES - 32));
    float v1 = has2 ? r[32 + lane] : -INFINITY;                 // cols 32..39

    float m = fmaxf(v0, v1);
    #pragma unroll
    for (int off = 16; off > 0; off >>= 1)
        m = fmaxf(m, __shfl_xor(m, off, 32));

    float s = expf(v0 - m) + (has2 ? expf(v1 - m) : 0.0f);
    #pragma unroll
    for (int off = 16; off > 0; off >>= 1)
        s += __shfl_xor(s, off, 32);

    float lse = m + logf(s);
    r[lane] = v0 - lse;
    if (has2) r[32 + lane] = v1 - lse;
}

// ---------------- launcher ----------------

extern "C" void kernel_launch(void* const* d_in, const int* in_sizes, int n_in,
                              void* d_out, int out_size, void* d_ws, size_t ws_size,
                              hipStream_t stream) {
    const float* x  = (const float*)d_in[0];   // [N, 50]
    const int*   ei = (const int*)  d_in[1];   // [2, E]
    const float* W  = (const float*)d_in[2];   // [50, 40]
    const float* b  = (const float*)d_in[3];   // [40]
    float* out = (float*)d_out;                // [N, 40]

    float* ws    = (float*)d_ws;
    float* dinv  = ws;                                        // N floats
    float* h0    = ws + N_NODES;                              // N*50 floats
    float* h1p   = h0 + (size_t)N_NODES * N_FEAT;             // N*64 floats (padded)
    _Float16* bpack = (_Float16*)(h1p + (size_t)N_NODES * HSTRIDE);  // 6*32*16 f16 (6 KB)

    const int* row = ei;              // edge_index[0] : sources
    const int* col = ei + N_EDGES;    // edge_index[1] : targets

    // 1) degrees + D^{-1/2}
    k_deg_init<<<(N_NODES + 255) / 256, 256, 0, stream>>>(dinv);
    k_deg_accum<<<2048, 256, 0, stream>>>(col, dinv);
    k_rsqrt_inplace<<<(N_NODES + 255) / 256, 256, 0, stream>>>(dinv);

    // pre-pack W into WMMA B fragments (tiny, overlaps nothing critical)
    k_pack_B<<<1, 192, 0, stream>>>(W, bpack);

    // 2) hop 1: x -> h0 (stride 50)
    k_selfloop50<<<(N_NODES * N_FEAT + 255) / 256, 256, 0, stream>>>(x, dinv, h0);
    k_edge_hop<N_FEAT><<<4096, 256, 0, stream>>>(row, col, dinv, x, h0);

    //    hop 2: h0 -> h1p (stride 64, zero-padded cols 50..63)
    k_selfloop_pad<<<(N_NODES * HSTRIDE + 255) / 256, 256, 0, stream>>>(h0, dinv, h1p);
    k_edge_hop<HSTRIDE><<<4096, 256, 0, stream>>>(row, col, dinv, h0, h1p);

    // 3) logits = h1 @ W + b   (WMMA; 6250 row-tiles, 8 waves/block)
    k_linear_wmma<<<(N_NODES / 16 + 7) / 8, 256, 0, stream>>>(h1p, (const v16h*)bpack, b, out);

    // 4) log-softmax in place (one wave per row)
    k_logsoftmax<<<(N_NODES + 7) / 8, 256, 0, stream>>>(out);
}
